// MLPDecoder_26877905339092
// MI455X (gfx1250) — compile-verified, matrix operand order checked by reading, not run on hardware
//
#include <hip/hip_runtime.h>

typedef __attribute__((ext_vector_type(16))) _Float16 v16h;
typedef __attribute__((ext_vector_type(8)))  _Float16 v8h;
typedef __attribute__((ext_vector_type(4)))  _Float16 v4h;
typedef __attribute__((ext_vector_type(8)))  float    v8f;
typedef __attribute__((ext_vector_type(4)))  float    v4f;

#define DIM 256
#define ROWPAD 264            // 256 halves + 8-half (16 B) pad: keeps 16B alignment, breaks bank conflicts

// ---------------------------------------------------------------------------
// Prep: convert a 256x256 fp32 weight matrix (row-major [K][N]) into f16,
// pre-swizzled into per-lane WMMA B-fragment order:
//   linear = ((n_tile*8 + k_step)*32 + lane)*16 + r*2 + h
// B-fragment element (ISA 7.12.2, 16-bit layout):
//   K = k*32 + (lane>=16 ? 8 : 0) + (r<4 ? 2r+h : 16 + 2(r-4)+h)
//   N = n*16 + (lane & 15)
// ---------------------------------------------------------------------------
__global__ __launch_bounds__(256)
void weight_swizzle_f16_kernel(const float* __restrict__ W, _Float16* __restrict__ out) {
    int tid = blockIdx.x * 256 + threadIdx.x;        // 65536 total
    int h    =  tid        & 1;
    int r    = (tid >> 1)  & 7;
    int lane = (tid >> 4)  & 31;
    int k    = (tid >> 9)  & 7;
    int n    = (tid >> 12) & 15;
    int kk = (r < 4) ? (2 * r + h) : (16 + 2 * (r - 4) + h);
    int K = k * 32 + ((lane >> 4) << 3) + kk;
    int N = n * 16 + (lane & 15);
    out[tid] = (_Float16)W[K * DIM + N];
}

__device__ __forceinline__ v16h load_a_frag(const _Float16* ap) {
    v8h lo = *(const v8h*)(ap);
    v8h hi = *(const v8h*)(ap + 16);
    return __builtin_shufflevector(lo, hi, 0,1,2,3,4,5,6,7,8,9,10,11,12,13,14,15);
}

__device__ __forceinline__ v8f wmma_f16(v16h a, v16h b, v8f c) {
    return __builtin_amdgcn_wmma_f32_16x16x32_f16(false, a, false, b, (short)0, c, false, false);
}

// ---------------------------------------------------------------------------
// Main: one persistent wave32 per block; each wave loops over 32-edge tiles.
// Per tile (M = 32 rows = 2 row-tiles, so each B fragment feeds 2 WMMAs):
//   e  = x[src]*x[dst]  (f32 gather, L2-resident) -> f16 LDS tile [32 x 256]
//   h1 = relu(e @ W1 + b1)     256 v_wmma_f32_16x16x32_f16 (dynamic)
//   h2 = relu(h1 @ W2 + b2)    256 v_wmma (layer-3 dot fused per N-half)
//   out = h2 @ W3 + b3         shfl_xor reduce over each 16-lane half
// N in two rolled halves of 8 tiles (128 acc VGPRs live). All 8 B fragments
// of a k-step are preloaded into a register array so 16 global_load_b128 are
// in flight while the 16 WMMAs drain (no full-drain s_wait per WMMA pair).
// Accumulators start at 0 (inline SRC2=0 on first k-step); bias folds into
// the epilogue.
// ---------------------------------------------------------------------------
__global__ __launch_bounds__(32)
void edge_mlp_kernel(const float* __restrict__ x,
                     const int*   __restrict__ pos_src, const int* __restrict__ pos_dst,
                     const int*   __restrict__ neg_src, const int* __restrict__ neg_dst,
                     const _Float16* __restrict__ w1f, const float* __restrict__ b1,
                     const _Float16* __restrict__ w2f, const float* __restrict__ b2,
                     const float* __restrict__ w3,  const float* __restrict__ b3,
                     float* __restrict__ out, int tilesPerSet)
{
    __shared__ __align__(16) _Float16 ebuf [32][ROWPAD];
    __shared__ __align__(16) _Float16 h1buf[32][ROWPAD];

    const int lane = threadIdx.x & 31;
    const int row  = lane & 15;        // matrix row (M) within a 16-row tile
    const int hf   = lane >> 4;        // half-wave select
    const int col  = lane & 15;        // C/D column within a 16x16 tile
    const int klo  = hf << 3;          // A-fragment K offset (0 or 8)

    const float bb = b3[0];
    const int totalTiles = 2 * tilesPerSet;

    #pragma unroll 1
    for (int tile = blockIdx.x; tile < totalTiles; tile += gridDim.x) {
        const int  set   = (tile >= tilesPerSet) ? 1 : 0;
        const int  local = set ? (tile - tilesPerSet) : tile;
        const int* sidx  = set ? neg_src : pos_src;
        const int* didx  = set ? neg_dst : pos_dst;

        // prefetch next tile's edge-index cachelines (global_prefetch_b8)
        {
            const int nt = tile + gridDim.x;
            if (nt < totalTiles) {
                const int  ns = (nt >= tilesPerSet) ? 1 : 0;
                const int  nl = ns ? (nt - tilesPerSet) : nt;
                const int* psi = ns ? neg_src : pos_src;
                const int* pdi = ns ? neg_dst : pos_dst;
                __builtin_prefetch(&psi[nl * 32 + lane], 0, 1);
                __builtin_prefetch(&pdi[nl * 32 + lane], 0, 1);
            }
        }

        // ---- gather + elementwise product -> f16 LDS tile (32 rows) ----
        #pragma unroll
        for (int m = 0; m < 2; ++m) {
            const int  edge = local * 32 + m * 16 + row;
            const long s = sidx[edge];
            const long d = didx[edge];
            const v4f* xs = (const v4f*)(x + s * (long)DIM + hf * 128);
            const v4f* xd = (const v4f*)(x + d * (long)DIM + hf * 128);
            _Float16* erow = &ebuf[m * 16 + row][hf * 128];
            #pragma unroll 8
            for (int c = 0; c < 32; ++c) {
                v4f p = xs[c] * xd[c];
                *(v4h*)(erow + c * 4) = __builtin_convertvector(p, v4h);
            }
        }
        asm volatile("s_wait_dscnt 0" ::: "memory");   // cross-lane LDS handoff

        // ------------------------------ layer 1 ------------------------------
        #pragma unroll 1
        for (int nh = 0; nh < 2; ++nh) {
            const _Float16* wbase = w1f + (size_t)nh * (8 * 8 * 32 * 16);
            v8f acc[2][8];
            #pragma unroll
            for (int n = 0; n < 8; ++n) {
                acc[0][n] = (v8f){0.f, 0.f, 0.f, 0.f, 0.f, 0.f, 0.f, 0.f};
                acc[1][n] = (v8f){0.f, 0.f, 0.f, 0.f, 0.f, 0.f, 0.f, 0.f};
            }
            #pragma unroll
            for (int k = 0; k < 8; ++k) {
                const v16h a0 = load_a_frag(&ebuf[row     ][k * 32 + klo]);
                const v16h a1 = load_a_frag(&ebuf[row + 16][k * 32 + klo]);
                v16h bfrag[8];
                #pragma unroll
                for (int n = 0; n < 8; ++n)
                    bfrag[n] = *(const v16h*)(wbase + (((n * 8 + k) * 32 + lane) << 4));
                #pragma unroll
                for (int n = 0; n < 8; ++n) {
                    acc[0][n] = wmma_f16(a0, bfrag[n], acc[0][n]);
                    acc[1][n] = wmma_f16(a1, bfrag[n], acc[1][n]);
                }
            }
            // bias + relu + store h1 to second LDS buffer (ebuf still live for nh=1)
            #pragma unroll
            for (int n = 0; n < 8; ++n) {
                const int   ng = nh * 8 + n;
                const float bv = b1[ng * 16 + col];
                #pragma unroll
                for (int m = 0; m < 2; ++m) {
                    #pragma unroll
                    for (int r = 0; r < 8; ++r) {
                        float v = acc[m][n][r] + bv;
                        v = v > 0.f ? v : 0.f;
                        h1buf[m * 16 + r + hf * 8][ng * 16 + col] = (_Float16)v;
                    }
                }
            }
        }
        asm volatile("s_wait_dscnt 0" ::: "memory");   // cross-lane LDS handoff

        // ----------------------- layer 2 (+ fused layer 3) -----------------------
        float p[2][8];
        #pragma unroll
        for (int r = 0; r < 8; ++r) { p[0][r] = 0.f; p[1][r] = 0.f; }

        #pragma unroll 1
        for (int nh = 0; nh < 2; ++nh) {
            const _Float16* wbase = w2f + (size_t)nh * (8 * 8 * 32 * 16);
            v8f acc[2][8];
            #pragma unroll
            for (int n = 0; n < 8; ++n) {
                acc[0][n] = (v8f){0.f, 0.f, 0.f, 0.f, 0.f, 0.f, 0.f, 0.f};
                acc[1][n] = (v8f){0.f, 0.f, 0.f, 0.f, 0.f, 0.f, 0.f, 0.f};
            }
            #pragma unroll
            for (int k = 0; k < 8; ++k) {
                const v16h a0 = load_a_frag(&h1buf[row     ][k * 32 + klo]);
                const v16h a1 = load_a_frag(&h1buf[row + 16][k * 32 + klo]);
                v16h bfrag[8];
                #pragma unroll
                for (int n = 0; n < 8; ++n)
                    bfrag[n] = *(const v16h*)(wbase + (((n * 8 + k) * 32 + lane) << 4));
                #pragma unroll
                for (int n = 0; n < 8; ++n) {
                    acc[0][n] = wmma_f16(a0, bfrag[n], acc[0][n]);
                    acc[1][n] = wmma_f16(a1, bfrag[n], acc[1][n]);
                }
            }
            // bias + relu + fold this N-half of h2 into the layer-3 dot product
            #pragma unroll
            for (int n = 0; n < 8; ++n) {
                const int   ng = nh * 8 + n;
                const float bv = b2[ng * 16 + col];
                const float w  = w3[ng * 16 + col];
                #pragma unroll
                for (int m = 0; m < 2; ++m) {
                    #pragma unroll
                    for (int r = 0; r < 8; ++r) {
                        float v = acc[m][n][r] + bv;
                        v = v > 0.f ? v : 0.f;
                        p[m][r] += v * w;
                    }
                }
            }
        }

        // reduce across the 16 lanes of each half-wave (stays within half: masks 1,2,4,8)
        #pragma unroll
        for (int m = 0; m < 2; ++m) {
            #pragma unroll
            for (int r = 0; r < 8; ++r) {
                p[m][r] += __shfl_xor(p[m][r], 1, 32);
                p[m][r] += __shfl_xor(p[m][r], 2, 32);
                p[m][r] += __shfl_xor(p[m][r], 4, 32);
                p[m][r] += __shfl_xor(p[m][r], 8, 32);
            }
        }
        if (col == 0) {
            const long base = ((long)set * tilesPerSet + local) * 32;
            #pragma unroll
            for (int m = 0; m < 2; ++m) {
                #pragma unroll
                for (int r = 0; r < 8; ++r) {
                    out[base + m * 16 + hf * 8 + r] = p[m][r] + bb;
                }
            }
        }
    }
}

// ---------------------------------------------------------------------------
extern "C" void kernel_launch(void* const* d_in, const int* in_sizes, int n_in,
                              void* d_out, int out_size, void* d_ws, size_t ws_size,
                              hipStream_t stream) {
    const float* x       = (const float*)d_in[0];
    const int*   pos_src = (const int*)d_in[1];
    const int*   pos_dst = (const int*)d_in[2];
    const int*   neg_src = (const int*)d_in[3];
    const int*   neg_dst = (const int*)d_in[4];
    const float* W1      = (const float*)d_in[5];
    const float* b1      = (const float*)d_in[6];
    const float* W2      = (const float*)d_in[7];
    const float* b2      = (const float*)d_in[8];
    const float* W3      = (const float*)d_in[9];
    const float* b3      = (const float*)d_in[10];

    _Float16* w1f = (_Float16*)d_ws;            // 65536 halves = 128 KB
    _Float16* w2f = w1f + DIM * DIM;            // next 128 KB

    // swizzle weights into WMMA B-fragment order (f16)
    weight_swizzle_f16_kernel<<<DIM * DIM / 256, 256, 0, stream>>>(W1, w1f);
    weight_swizzle_f16_kernel<<<DIM * DIM / 256, 256, 0, stream>>>(W2, w2f);

    const int E = in_sizes[1];                  // 200000
    const int tilesPerSet = E / 32;             // 6250 tiles of 32 edges
    const int totalTiles  = 2 * tilesPerSet;    // 12500
    int blocks = totalTiles / 4;                // persistent waves, ~4 tiles each
    if (blocks < 1) blocks = 1;

    edge_mlp_kernel<<<blocks, 32, 0, stream>>>(
        x, pos_src, pos_dst, neg_src, neg_dst,
        w1f, b1, w2f, b2, W3, b3,
        (float*)d_out, tilesPerSet);
}
